// MemoryAttention_26749056319717
// MI455X (gfx1250) — matrix operand order repaired and checked by reading, not compiled, not run on hardware
//
#include <hip/hip_runtime.h>

// ---------------------------------------------------------------------------
// MemoryAttention for MI455X (gfx1250, wave32, WMMA).
// Roofline: 256 MB attention_weights stream dominates (≈15 us @ 23.3 TB/s);
// the 31 GFLOP of GEMM work runs on the bf16 WMMA path (f32 accumulate) so the
// kernel stays memory-bound. Softmax statistics are fp32. K/V tiles are staged
// block-wide through LDS with global_load_async_to_lds_b128 (ASYNCcnt) since
// all 8 waves of an attention block share the same (batch, head).
// ---------------------------------------------------------------------------

namespace {
constexpr int kB = 8, kH = 8, kNQ = 1024, kNK = 1024, kDM = 512;
constexpr int kDK = 64, kDV = 64, kM = 40;
constexpr int kNKM = kNK + kM;     // 1064 valid keys
constexpr int kNPAD = 1088;        // padded to 34 chunks of 32 keys
constexpr int kNCH = kNPAD / 32;   // 34
}

typedef __attribute__((ext_vector_type(16))) __bf16 bf16x16;
typedef __attribute__((ext_vector_type(8)))  __bf16 bf16x8;
typedef __attribute__((ext_vector_type(8)))  float  f32x8;
typedef __attribute__((ext_vector_type(4)))  float  f32x4;
typedef __attribute__((address_space(3))) __bf16 lds_bf16_t;

static __device__ __forceinline__ bf16x16 join16(bf16x8 lo, bf16x8 hi) {
  bf16x16 r;
#pragma unroll
  for (int e = 0; e < 8; ++e) { r[e] = lo[e]; r[8 + e] = hi[e]; }
  return r;
}

static __device__ __forceinline__ f32x8 zero8() {
  f32x8 r = {0.f, 0.f, 0.f, 0.f, 0.f, 0.f, 0.f, 0.f};
  return r;
}

static __device__ __forceinline__ f32x8 wmma_bf16(bf16x16 a, bf16x16 b, f32x8 c) {
  // D = A(16x32 bf16) * B(32x16 bf16) + C(16x16 f32)
  return __builtin_amdgcn_wmma_f32_16x16x32_bf16(false, a, false, b,
                                                 (short)0, c, false, false);
}

// Async DMA of one 16-byte chunk global -> LDS (tracked by ASYNCcnt).
static __device__ __forceinline__ void async_ld16(const __bf16* g, __bf16* l) {
  unsigned lds  = (unsigned)(unsigned long long)(lds_bf16_t*)l;  // LDS byte addr
  unsigned long long ga = (unsigned long long)g;
  asm volatile("global_load_async_to_lds_b128 %0, %1, off"
               :: "v"(lds), "v"(ga) : "memory");
}

static __device__ __forceinline__ void wait_async0() {
  asm volatile("s_wait_asynccnt 0" ::: "memory");
}

// P(16x32) * V(32x16) accumulate; vp points at LDS V^T row chunk (contig keys)
static __device__ __forceinline__ f32x8 pv_step(bf16x16 ap, const __bf16* vp, f32x8 acc) {
  bf16x16 bv = join16(*(const bf16x8*)vp, *(const bf16x8*)(vp + 8));
  return wmma_bf16(ap, bv, acc);
}

// ---------------------------------------------------------------------------
// Kernel 1: Wt[p][n][k] = W_p[k][n] as bf16 (B-fragments become contiguous)
// ---------------------------------------------------------------------------
extern "C" __global__ __launch_bounds__(256) void ma_prep_weights(
    const float* __restrict__ Wq, const float* __restrict__ Wk,
    const float* __restrict__ Wv, __bf16* __restrict__ Wt) {
  int t = blockIdx.x * 256 + threadIdx.x;   // [p][n][k], 3*512*512 threads
  int k = t & 511;
  int n = (t >> 9) & 511;
  int p = t >> 18;
  const float* W = (p == 0) ? Wq : (p == 1) ? Wk : Wv;
  Wt[t] = (__bf16)W[(size_t)k * kDM + n];
}

// ---------------------------------------------------------------------------
// Kernel 2: memory slots (sqrt(dk)*mem_keys) + zero padding into K and V^T
// ---------------------------------------------------------------------------
extern "C" __global__ __launch_bounds__(256) void ma_fill_mem(
    const float* __restrict__ memk, __bf16* __restrict__ Kb,
    __bf16* __restrict__ Vt) {
  int t = blockIdx.x * 256 + threadIdx.x;   // b(8) h(8) s(64) d(64)
  int d = t & 63;
  int s = (t >> 6) & 63;
  int h = (t >> 12) & 7;
  int b = t >> 15;
  float val = (s < kM) ? 8.0f * memk[(size_t)s * (kH * kDK) + h * kDK + d] : 0.0f;
  __bf16 w = (__bf16)val;
  size_t bh = (size_t)b * kH + h;
  int key = kNK + s;
  Kb[(bh * kNPAD + key) * kDK + d] = w;
  Vt[(bh * kDV + d) * kNPAD + key] = w;
}

// ---------------------------------------------------------------------------
// Kernel 3: fused QKV projection, one 16x16 output tile per wave, K=512
// Q -> [b,h,q,dk], K -> [b,h,key,dk], V -> [b,h,dv,key] (transposed)
// ---------------------------------------------------------------------------
extern "C" __global__ __launch_bounds__(256) void ma_proj(
    const float* __restrict__ Xq, const float* __restrict__ Xk,
    const float* __restrict__ Xv, const float* __restrict__ bq,
    const float* __restrict__ bk, const float* __restrict__ bvp,
    const __bf16* __restrict__ Wt, __bf16* __restrict__ Qb,
    __bf16* __restrict__ Kb, __bf16* __restrict__ Vt) {
  int wave = (blockIdx.x * 256 + threadIdx.x) >> 5;
  int lane = threadIdx.x & 31;
  int p = wave >> 14;             // 16384 tiles (512 row x 32 col) per proj
  int rem = wave & 16383;
  int rt = rem >> 5;              // 16-row tile of the 8192-row X
  int ct = rem & 31;              // 16-col tile of the 512 output cols
  int n  = lane & 15;
  int hi = lane >> 4;

  const float* X    = (p == 0) ? Xq : (p == 1) ? Xk : Xv;
  const float* bias = (p == 0) ? bq : (p == 1) ? bk : bvp;
  int colg = ct * 16 + n;
  const __bf16* wcol = Wt + ((size_t)p * kDM + colg) * kDM;
  const float*  arow = X + (size_t)(rt * 16 + n) * kDM + 8 * hi;

  f32x8 acc = zero8();
  for (int kk = 0; kk < kDM; kk += 32) {
    // A-fragment: row m = lane%16; K chunks [8hi..8hi+7] and [16+8hi..+7]
    const f32x4* apf = (const f32x4*)(arow + kk);
    f32x4 x0 = apf[0], x1 = apf[1], x2 = apf[4], x3 = apf[5];
    bf16x16 a, b;
#pragma unroll
    for (int e = 0; e < 4; ++e) {
      a[e]      = (__bf16)x0[e];
      a[4 + e]  = (__bf16)x1[e];
      a[8 + e]  = (__bf16)x2[e];
      a[12 + e] = (__bf16)x3[e];
    }
    // B-fragment: col n = lane%16; K chunks [8hi..8hi+7] and [16+8hi..+7]
    const bf16x8* bp = (const bf16x8*)(wcol + kk + 8 * hi);
    bf16x8 w0 = bp[0], w1 = bp[2];
#pragma unroll
    for (int e = 0; e < 8; ++e) { b[e] = w0[e]; b[8 + e] = w1[e]; }
    acc = wmma_bf16(a, b, acc);
  }

  float bi = bias[colg];
  int h = colg >> 6, d = colg & 63;   // output feature -> (head, dk/dv)
  if (p == 2) {
    // V transposed: lane's 8 C rows are 8 consecutive keys -> one b128 store
    int rr0 = rt * 16 + 8 * hi;
    int bb = rr0 >> 10, key0 = rr0 & 1023;
    bf16x8 pk;
#pragma unroll
    for (int i = 0; i < 8; ++i) pk[i] = (__bf16)(acc[i] + bi);
    *(bf16x8*)(Vt + (((size_t)bb * kH + h) * kDV + d) * kNPAD + key0) = pk;
  } else {
    __bf16* dst = (p == 0) ? Qb : Kb;
    int rowlen = (p == 0) ? kNQ : kNPAD;
#pragma unroll
    for (int i = 0; i < 8; ++i) {
      int rr = rt * 16 + i + 8 * hi;      // C row m = i + 8*(lane>=16)
      int bb = rr >> 10, ri = rr & 1023;
      dst[(((size_t)bb * kH + h) * rowlen + ri) * kDK + d] = (__bf16)(acc[i] + bi);
    }
  }
}

// ---------------------------------------------------------------------------
// Kernel 4: flash attention. One block = 8 waves sharing (b,h), 128 q rows.
// Per 32-key chunk: async-DMA K(32x64) and V^T(64x32) tiles into LDS once for
// the whole block, then each wave runs 4 S-WMMAs + online softmax + 4 PV-WMMAs.
// ---------------------------------------------------------------------------
extern "C" __global__ __launch_bounds__(256) void ma_attn(
    const __bf16* __restrict__ Qb, const __bf16* __restrict__ Kb,
    const __bf16* __restrict__ Vt, const float* __restrict__ AW,
    float* __restrict__ out) {
  __shared__ __align__(16) __bf16 kbuf[32][72];     // key-major, pad -> no conflicts
  __shared__ __align__(16) __bf16 vbuf[64][40];     // dv-major,  pad -> no conflicts
  __shared__ __align__(16) __bf16 pbuf[8][16][40];  // per-wave P staging

  int tid  = threadIdx.x;
  int lane = tid & 31;
  int wid  = tid >> 5;
  int wave = blockIdx.x * 8 + wid;    // 4096 waves total
  int qt = wave & 63;
  int h  = (wave >> 6) & 7;           // uniform across the block
  int b  = wave >> 9;                 // uniform across the block
  int q0 = qt << 4;
  int n  = lane & 15;
  int hi = lane >> 4;
  size_t bh = (size_t)b * kH + h;

  // Block-wide staging assignment: 256 threads x one 16B chunk for K and V.
  int kkey = tid >> 3, kc = (tid & 7) * 8;   // K: 32 rows x 8 chunks of 8 bf16
  int vdv  = tid >> 2, vc = (tid & 3) * 8;   // V: 64 rows x 4 chunks of 8 bf16
  const __bf16* kgbase = Kb + (bh * kNPAD + kkey) * kDK + kc;  // + j0*kDK / iter
  const __bf16* vgbase = Vt + (bh * kDV + vdv) * kNPAD + vc;   // + j0     / iter

  // Q A-fragments (dk = 64 -> two K=32 fragments), resident for whole loop
  const __bf16* qrow = Qb + (bh * kNQ + q0 + n) * kDK;
  bf16x16 aq0 = join16(*(const bf16x8*)(qrow + 8 * hi),
                       *(const bf16x8*)(qrow + 16 + 8 * hi));
  bf16x16 aq1 = join16(*(const bf16x8*)(qrow + 32 + 8 * hi),
                       *(const bf16x8*)(qrow + 48 + 8 * hi));

  float mrun[8], lrun[8];
#pragma unroll
  for (int i = 0; i < 8; ++i) { mrun[i] = -1e30f; lrun[i] = 0.f; }
  f32x8 o0 = zero8(), o1 = zero8(), o2 = zero8(), o3 = zero8();

  const float* awq = AW + (bh * kNQ + q0) * kNK;   // [q][key] row-major

  for (int jc = 0; jc < kNCH; ++jc) {
    int j0 = jc * 32;

    // ---- block-wide async K/V tile fill (ASYNCcnt + workgroup barrier) ----
    __syncthreads();                               // prior consumers done
    async_ld16(kgbase + (size_t)j0 * kDK, &kbuf[kkey][kc]);
    async_ld16(vgbase + j0,               &vbuf[vdv][vc]);
    wait_async0();                                 // this wave's DMAs landed
    __syncthreads();                               // every wave's DMAs landed

    float sv[16];   // two 16-col score tiles, element i = C row i + 8*hi
#pragma unroll
    for (int t2 = 0; t2 < 2; ++t2) {
      int keyl = 16 * t2 + n;          // this lane's score column (chunk-local)
      const __bf16* krow = &kbuf[keyl][8 * hi];
      bf16x16 bk0 = join16(*(const bf16x8*)(krow),      *(const bf16x8*)(krow + 16));
      bf16x16 bk1 = join16(*(const bf16x8*)(krow + 32), *(const bf16x8*)(krow + 48));
      f32x8 c = zero8();
      c = wmma_bf16(aq0, bk0, c);
      c = wmma_bf16(aq1, bk1, c);
      int key = j0 + keyl;
      if (jc < 32) {  // real keys: scale + elementwise attention_weights stream
        const float* awp = awq + key + (size_t)(8 * hi) * kNK;
#pragma unroll
        for (int i = 0; i < 8; ++i) {
          float aww = __builtin_nontemporal_load(awp + (size_t)i * kNK);
          sv[t2 * 8 + i] = c[i] * 0.125f * aww;
        }
      } else {        // memory slots (no weights); mask padded keys
#pragma unroll
        for (int i = 0; i < 8; ++i)
          sv[t2 * 8 + i] = (key < kNKM) ? c[i] * 0.125f : -1e30f;
      }
    }
    if (jc + 1 < 32)  // hint the streaming-weight prefetcher for next chunk
      __builtin_prefetch(awq + (size_t)(8 * hi) * kNK + j0 + 32 + n, 0, 0);

    // Online softmax: row reductions across the 16-lane half via shfl_xor
    float alpha[8];
#pragma unroll
    for (int i = 0; i < 8; ++i) {
      float rm = fmaxf(sv[i], sv[8 + i]);
      rm = fmaxf(rm, __shfl_xor(rm, 1, 32));
      rm = fmaxf(rm, __shfl_xor(rm, 2, 32));
      rm = fmaxf(rm, __shfl_xor(rm, 4, 32));
      rm = fmaxf(rm, __shfl_xor(rm, 8, 32));
      float mnew = fmaxf(mrun[i], rm);
      float al = __expf(mrun[i] - mnew);
      mrun[i] = mnew;
      float p0 = __expf(sv[i] - mnew);
      float p1 = __expf(sv[8 + i] - mnew);
      sv[i] = p0; sv[8 + i] = p1;
      float rs = p0 + p1;
      rs += __shfl_xor(rs, 1, 32);
      rs += __shfl_xor(rs, 2, 32);
      rs += __shfl_xor(rs, 4, 32);
      rs += __shfl_xor(rs, 8, 32);
      lrun[i] = lrun[i] * al + rs;
      alpha[i] = al;
    }

    // C-layout -> A-layout conversion of P through wave-private LDS
#pragma unroll
    for (int i = 0; i < 8; ++i) {
      int m = i + 8 * hi;
      pbuf[wid][m][n]      = (__bf16)sv[i];
      pbuf[wid][m][16 + n] = (__bf16)sv[8 + i];
    }
    // DS ops are in-order within a wave: load after store is safe, no barrier
    bf16x16 apf = join16(*(const bf16x8*)&pbuf[wid][n][16 * hi],
                         *(const bf16x8*)&pbuf[wid][n][16 * hi + 8]);

#pragma unroll
    for (int i = 0; i < 8; ++i) {
      o0[i] *= alpha[i]; o1[i] *= alpha[i];
      o2[i] *= alpha[i]; o3[i] *= alpha[i];
    }
    o0 = pv_step(apf, &vbuf[n][16 * hi],      o0);
    o1 = pv_step(apf, &vbuf[16 + n][16 * hi], o1);
    o2 = pv_step(apf, &vbuf[32 + n][16 * hi], o2);
    o3 = pv_step(apf, &vbuf[48 + n][16 * hi], o3);
  }

  // Normalize and write fp32 output [b, q, h*64 + dv]
  float* orow = out + ((size_t)b * kNQ + q0) * (kH * kDV) + h * kDV;
#pragma unroll
  for (int i = 0; i < 8; ++i) {
    float inv = 1.0f / lrun[i];
    float* po = orow + (size_t)(i + 8 * hi) * (kH * kDV);
    po[n]      = o0[i] * inv;
    po[16 + n] = o1[i] * inv;
    po[32 + n] = o2[i] * inv;
    po[48 + n] = o3[i] * inv;
  }
}

// ---------------------------------------------------------------------------
extern "C" void kernel_launch(void* const* d_in, const int* in_sizes, int n_in,
                              void* d_out, int out_size, void* d_ws, size_t ws_size,
                              hipStream_t stream) {
  (void)in_sizes; (void)n_in; (void)out_size; (void)ws_size;
  const float* queries = (const float*)d_in[0];
  const float* keys    = (const float*)d_in[1];
  const float* values  = (const float*)d_in[2];
  const float* aw      = (const float*)d_in[3];
  const float* Wq = (const float*)d_in[4];
  const float* bq = (const float*)d_in[5];
  const float* Wk = (const float*)d_in[6];
  const float* bk = (const float*)d_in[7];
  const float* Wv = (const float*)d_in[8];
  const float* bv = (const float*)d_in[9];
  const float* mem = (const float*)d_in[10];

  char* ws = (char*)d_ws;
  size_t off = 0;
  auto take = [&](size_t bytes) {
    size_t o = (off + 255) & ~(size_t)255;
    off = o + bytes;
    return ws + o;
  };
  __bf16* Wt = (__bf16*)take(3ull * 512 * 512 * sizeof(short));              // 1.5 MB
  __bf16* Qb = (__bf16*)take((size_t)kB * kH * kNQ   * kDK * sizeof(short)); // 8 MB
  __bf16* Kb = (__bf16*)take((size_t)kB * kH * kNPAD * kDK * sizeof(short)); // 8.9 MB
  __bf16* Vt = (__bf16*)take((size_t)kB * kH * kDV * kNPAD * sizeof(short)); // 8.9 MB

  ma_prep_weights<<<3072, 256, 0, stream>>>(Wq, Wk, Wv, Wt);               // 3*512*512
  ma_fill_mem   <<<1024, 256, 0, stream>>>(mem, Kb, Vt);                   // 8*8*64*64
  ma_proj       <<<6144, 256, 0, stream>>>(queries, keys, values,
                                           bq, bk, bv, Wt, Qb, Kb, Vt);    // 49152 waves
  ma_attn       <<<512,  256, 0, stream>>>(Qb, Kb, Vt, aw, (float*)d_out); // 4096 waves
}
